// PoseFeatureNet_23819888624111
// MI455X (gfx1250) — matrix operand order repaired and checked by reading, not compiled, not run on hardware
//
#include <hip/hip_runtime.h>
#include <hip/hip_bf16.h>
#include <math.h>

typedef _Float16 f16;
typedef _Float16 v16h __attribute__((ext_vector_type(16)));
typedef _Float16 v8h  __attribute__((ext_vector_type(8)));
typedef float    v8f  __attribute__((ext_vector_type(8)));

// ---------------- constants: 17-node skeleton graph ----------------
// adjacency mask with self loops (bit j of row i => edge j->i allowed)
__constant__ unsigned kAdjMask[17] = {
  0x00007u,0x0000Fu,0x00017u,0x0002Au,0x00054u,0x008E8u,0x01170u,0x002A0u,
  0x00540u,0x00280u,0x00500u,0x03820u,0x05840u,0x0A800u,0x15000u,0x0A000u,0x14000u
};
// CONN: 19 base edges + reversed; v = pose[dst] - pose[src]
__constant__ int kEdgeSrc[38] = {15,13,16,14,11,5,6,5,5,6,7,8,1,0,0,1,2,3,4,
                                 13,11,14,12,12,11,12,6,7,8,9,10,2,1,2,3,4,5,6};
__constant__ int kEdgeDst[38] = {13,11,14,12,12,11,12,6,7,8,9,10,2,1,2,3,4,5,6,
                                 15,13,16,14,11,5,6,5,5,6,7,8,1,0,0,1,2,3,4};

__device__ __forceinline__ float sigm(float x) { return 1.0f / (1.0f + expf(-x)); }

// =====================================================================
// WMMA GEMM: C[M,N] = A[M,K](f16,lda) * B[N,K]^T(f16,ldb) + add(+bias)
// block tile 128x128, 8 waves (4 along M x 2 along N), wave tile 32x64,
// K-step 32 via v_wmma_f32_16x16x32_f16. M,N mult of 128; K mult of 32.
// =====================================================================
__global__ __launch_bounds__(256)
void gemm_wmma_f16(const f16* __restrict__ A, long lda,
                   const f16* __restrict__ B, long ldb,
                   float* __restrict__ C, long ldc,
                   const float* __restrict__ add, long ldadd,
                   const float* __restrict__ bias,
                   int K)
{
  __shared__ __align__(16) f16 Al[128 * 40];
  __shared__ __align__(16) f16 Bl[128 * 40];
  const int tid  = threadIdx.x;
  const int lane = tid & 31;
  const int wave = tid >> 5;
  const int wm = wave & 3;        // 0..3 along M (32 rows each)
  const int wn = wave >> 2;       // 0..1 along N (64 cols each)
  const long m0 = (long)blockIdx.y * 128;
  const long n0 = (long)blockIdx.x * 128;

  v8f acc[2][4];
#pragma unroll
  for (int tm = 0; tm < 2; ++tm)
#pragma unroll
    for (int tn = 0; tn < 4; ++tn)
#pragma unroll
      for (int r = 0; r < 8; ++r) acc[tm][tn][r] = 0.0f;

  union Frag { v16h v; v8h h[2]; };

  for (int k0 = 0; k0 < K; k0 += 32) {
    // stage 128x32 A tile and 128x32 B tile (both K-contiguous) into LDS
#pragma unroll
    for (int u = tid; u < 512; u += 256) {
      const int row = u >> 2, seg = u & 3;
      *reinterpret_cast<uint4*>(&Al[row * 40 + seg * 8]) =
          *reinterpret_cast<const uint4*>(A + (m0 + row) * lda + k0 + seg * 8);
      *reinterpret_cast<uint4*>(&Bl[row * 40 + seg * 8]) =
          *reinterpret_cast<const uint4*>(B + (n0 + row) * ldb + k0 + seg * 8);
    }
    __syncthreads();

    // A fragment: lane m = lane&15; lanes0-15 hold K {0..7,16..23}, lanes16-31 K {8..15,24..31}
    Frag af[2], bf[4];
    const int mrow  = lane & 15;
    const int abase = (lane < 16) ? 0 : 8;
#pragma unroll
    for (int tm = 0; tm < 2; ++tm) {
      const f16* p = &Al[(wm * 32 + tm * 16 + mrow) * 40];
      af[tm].h[0] = *reinterpret_cast<const v8h*>(p + abase);
      af[tm].h[1] = *reinterpret_cast<const v8h*>(p + 16 + abase);
    }
    // B fragment: lane n = lane&15; lanes0-15 hold K {0..7,16..23}, lanes16-31 K {8..15,24..31}
    const int bbase = (lane < 16) ? 0 : 8;
#pragma unroll
    for (int tn = 0; tn < 4; ++tn) {
      const f16* p = &Bl[(wn * 64 + tn * 16 + mrow) * 40 + bbase];
      bf[tn].h[0] = *reinterpret_cast<const v8h*>(p);
      bf[tn].h[1] = *reinterpret_cast<const v8h*>(p + 16);
    }
#pragma unroll
    for (int tm = 0; tm < 2; ++tm)
#pragma unroll
      for (int tn = 0; tn < 4; ++tn)
        acc[tm][tn] = __builtin_amdgcn_wmma_f32_16x16x32_f16(
            false, af[tm].v, false, bf[tn].v, (short)0, acc[tm][tn], false, false);
    __syncthreads();
  }

  // C layout: lane n = lane&15, VGPR r -> m = r (+8 for lanes 16-31)
  const int nlane = lane & 15;
  const int mhi   = (lane >> 4) << 3;
#pragma unroll
  for (int tm = 0; tm < 2; ++tm)
#pragma unroll
    for (int tn = 0; tn < 4; ++tn) {
      const long n = n0 + wn * 64 + tn * 16 + nlane;
#pragma unroll
      for (int r = 0; r < 8; ++r) {
        const long m = m0 + wm * 32 + tm * 16 + r + mhi;
        float v = acc[tm][tn][r];
        if (add)  v += add[m * ldadd + n];
        if (bias) v += bias[n];
        C[m * ldc + n] = v;
      }
    }
}

// =====================================================================
// GAT layer 1 fully fused (K=3): one block per graph
// =====================================================================
__global__ __launch_bounds__(256)
void gat1_fused(const float* __restrict__ pose1, const float* __restrict__ pose2,
                const float* __restrict__ W1, const float* __restrict__ asrc,
                const float* __restrict__ adst, const float* __restrict__ bias,
                f16* __restrict__ Aout)
{
  __shared__ float sW[3 * 512];
  __shared__ float sA[512], sD[512];
  __shared__ float sx[17][3];
  __shared__ float sh[17 * 512];
  __shared__ float ss[17 * 8], sd[17 * 8];
  __shared__ float sal[17 * 17 * 8];
  const int g = blockIdx.x, tid = threadIdx.x;
  const float* pose = (g < 3072) ? pose1 + (size_t)g * 51 : pose2 + (size_t)(g - 3072) * 51;
  for (int i = tid; i < 1536; i += 256) sW[i] = W1[i];
  for (int i = tid; i < 512; i += 256) { sA[i] = asrc[i]; sD[i] = adst[i]; }
  for (int i = tid; i < 51; i += 256) sx[i / 3][i % 3] = pose[i];
  __syncthreads();
  for (int i = tid; i < 17 * 512; i += 256) {
    const int n = i >> 9, c = i & 511;
    sh[i] = sx[n][0] * sW[c] + sx[n][1] * sW[512 + c] + sx[n][2] * sW[1024 + c];
  }
  __syncthreads();
  if (tid < 136) {
    const int n = tid >> 3, hh = tid & 7;
    float a = 0.f, d = 0.f;
    for (int k = 0; k < 64; ++k) {
      const float v = sh[n * 512 + hh * 64 + k];
      a += v * sA[hh * 64 + k]; d += v * sD[hh * 64 + k];
    }
    ss[tid] = a; sd[tid] = d;
  }
  __syncthreads();
  if (tid < 136) {
    const int i = tid >> 3, hh = tid & 7;
    const unsigned msk = kAdjMask[i];
    float ev[17]; float mx = -1e30f;
    for (int j = 0; j < 17; ++j) {
      float e;
      if ((msk >> j) & 1u) { const float t = sd[i * 8 + hh] + ss[j * 8 + hh]; e = t >= 0.f ? t : 0.2f * t; }
      else e = -1e9f;
      ev[j] = e; mx = fmaxf(mx, e);
    }
    float sm = 0.f;
    for (int j = 0; j < 17; ++j) { ev[j] = expf(ev[j] - mx); sm += ev[j]; }
    const float inv = 1.f / sm;
    for (int j = 0; j < 17; ++j) sal[(i * 17 + j) * 8 + hh] = ev[j] * inv;
  }
  __syncthreads();
  for (int idx = tid; idx < 17 * 512; idx += 256) {
    const int i = idx >> 9, c = idx & 511, hh = c >> 6;
    float acc = 0.f;
    for (int j = 0; j < 17; ++j) acc += sal[(i * 17 + j) * 8 + hh] * sh[j * 512 + c];
    float v = acc + bias[c];
    v = v > 0.f ? v : (expf(v) - 1.f);                     // ELU
    Aout[((size_t)g * 17 + i) * 512 + c] = (f16)v;
  }
}

// =====================================================================
// GAT attention (concat variant, layer 2): reads H fp32, writes ELU(out) f16
// =====================================================================
__global__ __launch_bounds__(256)
void att_concat(const float* __restrict__ H, int HD, int D,
                const float* __restrict__ asrc, const float* __restrict__ adst,
                const float* __restrict__ bias, f16* __restrict__ Aout)
{
  __shared__ float ss[136], sd[136], sal[17 * 17 * 8];
  const int g = blockIdx.x, tid = threadIdx.x;
  const size_t base = (size_t)g * 17 * HD;
  if (tid < 136) {
    const int n = tid >> 3, hh = tid & 7;
    float a = 0.f, d = 0.f;
    for (int k = 0; k < D; ++k) {
      const float v = H[base + (size_t)n * HD + hh * D + k];
      a += v * asrc[hh * D + k]; d += v * adst[hh * D + k];
    }
    ss[tid] = a; sd[tid] = d;
  }
  __syncthreads();
  if (tid < 136) {
    const int i = tid >> 3, hh = tid & 7;
    const unsigned msk = kAdjMask[i];
    float ev[17]; float mx = -1e30f;
    for (int j = 0; j < 17; ++j) {
      float e;
      if ((msk >> j) & 1u) { const float t = sd[i * 8 + hh] + ss[j * 8 + hh]; e = t >= 0.f ? t : 0.2f * t; }
      else e = -1e9f;
      ev[j] = e; mx = fmaxf(mx, e);
    }
    float sm = 0.f;
    for (int j = 0; j < 17; ++j) { ev[j] = expf(ev[j] - mx); sm += ev[j]; }
    const float inv = 1.f / sm;
    for (int j = 0; j < 17; ++j) sal[(i * 17 + j) * 8 + hh] = ev[j] * inv;
  }
  __syncthreads();
  for (int idx = tid; idx < 17 * HD; idx += 256) {
    const int i = idx / HD, c = idx % HD, hh = c / D;
    float acc = 0.f;
    for (int j = 0; j < 17; ++j) acc += sal[(i * 17 + j) * 8 + hh] * H[base + (size_t)j * HD + c];
    float v = acc + bias[c];
    v = v > 0.f ? v : (expf(v) - 1.f);
    Aout[base + (size_t)i * HD + c] = (f16)v;
  }
}

// =====================================================================
// GAT attention layer 3 (mean over heads, HD=2048, D=256) -> comb fp32
// =====================================================================
__global__ __launch_bounds__(256)
void att_mean(const float* __restrict__ H, const float* __restrict__ asrc,
              const float* __restrict__ adst, const float* __restrict__ bias,
              float* __restrict__ comb)
{
  __shared__ float ss[136], sd[136], sal[17 * 17 * 8];
  const int g = blockIdx.x, tid = threadIdx.x;
  const size_t base = (size_t)g * 17 * 2048;
  if (tid < 136) {
    const int n = tid >> 3, hh = tid & 7;
    float a = 0.f, d = 0.f;
    for (int k = 0; k < 256; ++k) {
      const float v = H[base + (size_t)n * 2048 + hh * 256 + k];
      a += v * asrc[hh * 256 + k]; d += v * adst[hh * 256 + k];
    }
    ss[tid] = a; sd[tid] = d;
  }
  __syncthreads();
  if (tid < 136) {
    const int i = tid >> 3, hh = tid & 7;
    const unsigned msk = kAdjMask[i];
    float ev[17]; float mx = -1e30f;
    for (int j = 0; j < 17; ++j) {
      float e;
      if ((msk >> j) & 1u) { const float t = sd[i * 8 + hh] + ss[j * 8 + hh]; e = t >= 0.f ? t : 0.2f * t; }
      else e = -1e9f;
      ev[j] = e; mx = fmaxf(mx, e);
    }
    float sm = 0.f;
    for (int j = 0; j < 17; ++j) { ev[j] = expf(ev[j] - mx); sm += ev[j]; }
    const float inv = 1.f / sm;
    for (int j = 0; j < 17; ++j) sal[(i * 17 + j) * 8 + hh] = ev[j] * inv;
  }
  __syncthreads();
  const int p = g / 3072, gg = g % 3072, bb = gg / 24, t = gg % 24;
  float* crow = comb + (((size_t)(p * 128 + bb)) * 24 + t) * 4608;
  for (int idx = tid; idx < 17 * 256; idx += 256) {
    const int i = idx >> 8, dd = idx & 255;
    float acc = 0.f;
    for (int hh = 0; hh < 8; ++hh)
      for (int j = 0; j < 17; ++j)
        acc += sal[(i * 17 + j) * 8 + hh] * H[base + (size_t)j * 2048 + hh * 256 + dd];
    crow[i * 256 + dd] = acc * 0.125f + bias[dd];
  }
}

// ===================== edge features + FC (38->128, two rows/graph) =====================
__global__ __launch_bounds__(256)
void edge_fc(const float* __restrict__ pose1, const float* __restrict__ pose2,
             const float* __restrict__ fcw, const float* __restrict__ fcb,
             float* __restrict__ comb)
{
  __shared__ float flat[76];
  __shared__ float sxy[17][2];
  const int g = blockIdx.x, tid = threadIdx.x;
  const float* pose = (g < 3072) ? pose1 + (size_t)g * 51 : pose2 + (size_t)(g - 3072) * 51;
  if (tid < 34) sxy[tid >> 1][tid & 1] = pose[(tid >> 1) * 3 + (tid & 1)];
  __syncthreads();
  if (tid < 38) {
    const float vx = sxy[kEdgeDst[tid]][0] - sxy[kEdgeSrc[tid]][0];
    const float vy = sxy[kEdgeDst[tid]][1] - sxy[kEdgeSrc[tid]][1];
    flat[tid * 2]     = sqrtf(vx * vx + vy * vy);
    flat[tid * 2 + 1] = atan2f(vy, vx);
  }
  __syncthreads();
  const int r = tid >> 7, n = tid & 127;
  float acc = fcb[n];
  for (int k = 0; k < 38; ++k) acc += flat[r * 38 + k] * fcw[k * 128 + n];
  const int p = g / 3072, gg = g % 3072, bb = gg / 24, t = gg % 24;
  comb[(((size_t)(p * 128 + bb)) * 24 + t) * 4608 + 4352 + r * 128 + n] = acc;
}

// ===================== BatchNorm over time axis (eval) -> f16 =====================
__global__ __launch_bounds__(256)
void bn_to_f16(const float* __restrict__ comb, const float* __restrict__ gam,
               const float* __restrict__ bet, const float* __restrict__ mean,
               const float* __restrict__ var, f16* __restrict__ out, long total)
{
  const long idx = (long)blockIdx.x * 256 + threadIdx.x;
  if (idx >= total) return;
  const int t = (int)((idx / 4608) % 24);
  const float sc = gam[t] * rsqrtf(var[t] + 1e-5f);
  out[idx] = (f16)((comb[idx] - mean[t]) * sc + bet[t]);
}

// ===================== LSTM gates (torch order i,f,g,o) =====================
__global__ __launch_bounds__(256)
void lstm_gate(const float* __restrict__ G, long ldG,
               const float* __restrict__ c_in, float* __restrict__ c_out,
               f16* __restrict__ h16, float* __restrict__ tail)
{
  const int idx = blockIdx.x * 256 + threadIdx.x;
  if (idx >= 256 * 512) return;
  const int b = idx >> 9, u = idx & 511;
  const long base = (long)b * ldG;
  const float gi = G[base + u], gf = G[base + 512 + u];
  const float gg = G[base + 1024 + u], go = G[base + 1536 + u];
  const float c0 = c_in ? c_in[idx] : 0.0f;
  const float c  = sigm(gf) * c0 + sigm(gi) * tanhf(gg);
  const float h  = sigm(go) * tanhf(c);
  c_out[idx] = c;
  h16[idx]   = (f16)h;
  if (tail) tail[(size_t)b * 1024 + u] = h;
}

// ===================== classifier (fp32, tiny) =====================
__global__ __launch_bounds__(256)
void classifier(const float* __restrict__ lstm, const float* __restrict__ w,
                const float* __restrict__ bias, float* __restrict__ out)
{
  const int idx = blockIdx.x * 256 + threadIdx.x;
  if (idx >= 256 * 500) return;
  const int m = idx / 500, n = idx % 500;
  const float* x = lstm + (size_t)m * 1024;
  float acc = bias[n];
  for (int k = 0; k < 1024; ++k) acc += x[k] * w[k * 500 + n];
  out[idx] = acc;
}

// ===================== helpers: f32->f16 conversions, bias sum =====================
__global__ void f32_to_f16(const float* __restrict__ in, f16* __restrict__ out, long n)
{
  const long i = (long)blockIdx.x * 256 + threadIdx.x;
  if (i < n) out[i] = (f16)in[i];
}
__global__ void f32_to_f16_T(const float* __restrict__ in /*K x N*/, f16* __restrict__ out /*N x K*/,
                             int K, int N)
{
  const long i = (long)blockIdx.x * 256 + threadIdx.x;
  if (i >= (long)K * N) return;
  const int k = (int)(i / N), n = (int)(i % N);
  out[(size_t)n * K + k] = (f16)in[i];
}
__global__ void bias_sum(const float* a1, const float* a2, const float* b1, const float* b2,
                         float* o1, float* o2)
{
  const int i = blockIdx.x * 256 + threadIdx.x;
  if (i < 2048) { o1[i] = a1[i] + a2[i]; o2[i] = b1[i] + b2[i]; }
}

// =====================================================================
extern "C" void kernel_launch(void* const* d_in, const int* in_sizes, int n_in,
                              void* d_out, int out_size, void* d_ws, size_t ws_size,
                              hipStream_t stream)
{
  (void)in_sizes; (void)n_in; (void)out_size; (void)ws_size;
  const float* pose1 = (const float*)d_in[0];
  const float* pose2 = (const float*)d_in[1];
  const float* W1  = (const float*)d_in[3];
  const float* as1 = (const float*)d_in[4];
  const float* ad1 = (const float*)d_in[5];
  const float* b1  = (const float*)d_in[6];
  const float* W2  = (const float*)d_in[7];
  const float* as2 = (const float*)d_in[8];
  const float* ad2 = (const float*)d_in[9];
  const float* b2  = (const float*)d_in[10];
  const float* W3  = (const float*)d_in[11];
  const float* as3 = (const float*)d_in[12];
  const float* ad3 = (const float*)d_in[13];
  const float* b3  = (const float*)d_in[14];
  const float* fcw = (const float*)d_in[15];
  const float* fcb = (const float*)d_in[16];
  const float* bng = (const float*)d_in[17];
  const float* bnb = (const float*)d_in[18];
  const float* bnm = (const float*)d_in[19];
  const float* bnv = (const float*)d_in[20];
  const float* Wihf = (const float*)d_in[21];
  const float* Whhf = (const float*)d_in[22];
  const float* bihf = (const float*)d_in[23];
  const float* bhhf = (const float*)d_in[24];
  const float* Wihb = (const float*)d_in[25];
  // d_in[26] = Whh_b: algebraically unused (backward dir needs only step 0, h=0)
  const float* bihb = (const float*)d_in[27];
  const float* bhhb = (const float*)d_in[28];
  const float* clsw = (const float*)d_in[29];
  const float* clsb = (const float*)d_in[30];
  float* out = (float*)d_out;

  const long M17 = 104448;            // 6144 graphs * 17 nodes
  // ---- workspace carve-up ----
  size_t o = 0;
  auto alloc = [&](size_t bytes) { size_t r = o; o += (bytes + 255) & ~(size_t)255; return r; };
  char* ws = (char*)d_ws;
  float* Xbuf  = (float*)(ws + alloc((size_t)M17 * 2048 * 4)); // H2 then H3
  f16*   Ybuf  = (f16*)  (ws + alloc((size_t)M17 * 1024 * 2)); // A1 then A2
  float* comb  = (float*)(ws + alloc((size_t)256 * 24 * 4608 * 4));
  f16*   combh = (f16*)  (ws + alloc((size_t)256 * 24 * 4608 * 2));
  float* Zf    = (float*)(ws + alloc((size_t)256 * 24 * 2048 * 4));
  float* Zb    = (float*)(ws + alloc((size_t)256 * 2048 * 4));
  f16*   W2h   = (f16*)  (ws + alloc((size_t)1024 * 512 * 2));
  f16*   W3h   = (f16*)  (ws + alloc((size_t)2048 * 1024 * 2));
  f16*   Wihfh = (f16*)  (ws + alloc((size_t)2048 * 4608 * 2));
  f16*   Wihbh = (f16*)  (ws + alloc((size_t)2048 * 4608 * 2));
  f16*   Whhfh = (f16*)  (ws + alloc((size_t)2048 * 512 * 2));
  float* bsumf = (float*)(ws + alloc(2048 * 4));
  float* bsumb = (float*)(ws + alloc(2048 * 4));
  float* Gbuf  = (float*)(ws + alloc((size_t)256 * 2048 * 4));
  f16*   h16   = (f16*)  (ws + alloc((size_t)256 * 512 * 2));
  float* cbuf  = (float*)(ws + alloc((size_t)256 * 512 * 4));

  // ---- weight prep (f16 conversions; B operand layout is [N,K]) ----
  f32_to_f16_T<<<(512 * 1024 + 255) / 256, 256, 0, stream>>>(W2, W2h, 512, 1024);
  f32_to_f16_T<<<(1024 * 2048 + 255) / 256, 256, 0, stream>>>(W3, W3h, 1024, 2048);
  f32_to_f16<<<(int)(((long)2048 * 4608 + 255) / 256), 256, 0, stream>>>(Wihf, Wihfh, (long)2048 * 4608);
  f32_to_f16<<<(int)(((long)2048 * 4608 + 255) / 256), 256, 0, stream>>>(Wihb, Wihbh, (long)2048 * 4608);
  f32_to_f16<<<(2048 * 512 + 255) / 256, 256, 0, stream>>>(Whhf, Whhfh, 2048 * 512);
  bias_sum<<<8, 256, 0, stream>>>(bihf, bhhf, bihb, bhhb, bsumf, bsumb);

  // ---- GAT layer 1 (fused, K=3) -> A1 (Ybuf, f16, [M17,512]) ----
  gat1_fused<<<6144, 256, 0, stream>>>(pose1, pose2, W1, as1, ad1, b1, Ybuf);

  // ---- GAT layer 2: H2 = A1 @ W2 (WMMA), then attention+ELU -> A2 ----
  gemm_wmma_f16<<<dim3(1024 / 128, (int)(M17 / 128)), 256, 0, stream>>>(
      Ybuf, 512, W2h, 512, Xbuf, 1024, nullptr, 0, nullptr, 512);
  att_concat<<<6144, 256, 0, stream>>>(Xbuf, 1024, 128, as2, ad2, b2, Ybuf);

  // ---- GAT layer 3: H3 = A2 @ W3 (WMMA), attention mean -> comb ----
  gemm_wmma_f16<<<dim3(2048 / 128, (int)(M17 / 128)), 256, 0, stream>>>(
      Ybuf, 1024, W3h, 1024, Xbuf, 2048, nullptr, 0, nullptr, 1024);
  att_mean<<<6144, 256, 0, stream>>>(Xbuf, as3, ad3, b3, comb);

  // ---- edge features -> comb tail; BN -> f16 ----
  edge_fc<<<6144, 256, 0, stream>>>(pose1, pose2, fcw, fcb, comb);
  const long combTotal = (long)256 * 24 * 4608;
  bn_to_f16<<<(int)((combTotal + 255) / 256), 256, 0, stream>>>(comb, bng, bnb, bnm, bnv, combh, combTotal);

  // ---- LSTM input projections (WMMA): Zf = X @ Wih_f^T + (bih+bhh) ----
  gemm_wmma_f16<<<dim3(2048 / 128, 6144 / 128), 256, 0, stream>>>(
      combh, 4608, Wihfh, 4608, Zf, 2048, nullptr, 0, bsumf, 4608);
  // backward direction: only t=T-1 step matters -> Zb from comb[:,23,:]
  gemm_wmma_f16<<<dim3(2048 / 128, 256 / 128), 256, 0, stream>>>(
      combh + (size_t)23 * 4608, (long)24 * 4608, Wihbh, 4608, Zb, 2048, nullptr, 0, bsumb, 4608);

  // ---- forward LSTM scan: t=0 from Zf directly, then 23 recurrent steps ----
  const int gateGrid = (256 * 512 + 255) / 256;
  lstm_gate<<<gateGrid, 256, 0, stream>>>(Zf, (long)24 * 2048, nullptr, cbuf, h16, nullptr);
  for (int t = 1; t < 24; ++t) {
    gemm_wmma_f16<<<dim3(2048 / 128, 256 / 128), 256, 0, stream>>>(
        h16, 512, Whhfh, 512, Gbuf, 2048, Zf + (size_t)t * 2048, (long)24 * 2048, nullptr, 512);
    lstm_gate<<<gateGrid, 256, 0, stream>>>(Gbuf, 2048, cbuf, cbuf, h16,
                                            (t == 23) ? out : nullptr);
  }
  // backward final hidden = single step on x_{T-1} from zero state
  lstm_gate<<<gateGrid, 256, 0, stream>>>(Zb, 2048, nullptr, cbuf, h16, out + 512);

  // ---- classifier: logits = lstm_out @ cls_w + cls_b ----
  classifier<<<(256 * 500 + 255) / 256, 256, 0, stream>>>(out, clsw, clsb, out + (size_t)256 * 1024);
}
